// GCNEncoderWithFeatures_12232066859182
// MI455X (gfx1250) — compile-verified, matrix operand order checked by reading, not run on hardware
//
#include <hip/hip_runtime.h>
#include <hip/hip_bf16.h>

// ---------------------------------------------------------------------------
// GCN encoder for MI455X (gfx1250, wave32).
// Dense GEMMs via v_wmma_f32_16x16x32_bf16 (fp32 accumulate), SpMM via
// float4 gather + global atomicAdd scatter (L2-resident working set).
// ---------------------------------------------------------------------------

#define N_NODES 50000
#define D_IN    256
#define H1      256
#define H2      128
#define H3      64
#define N_EDGES 800000

typedef __attribute__((ext_vector_type(16))) __bf16 bf16x16;
typedef __attribute__((ext_vector_type(8)))  __bf16 bf16x8;
typedef __attribute__((ext_vector_type(8)))  float  f32x8;

union Frag {
    bf16x16 v;
    bf16x8  h[2];
    __device__ Frag() {}
};

// Load one 16x32-bf16 WMMA operand fragment (A layout; B uses identical code
// on a pre-transposed [N][K] weight matrix).
// Lane l (<16):  elems 0..7 = row, K=k0..k0+7 ; elems 8..15 = K=k0+16..k0+23
// Lane l (>=16): same but K base shifted by +8 (handled via khalf).
__device__ __forceinline__ bf16x16 load_frag(const __bf16* __restrict__ base,
                                             int ld, int row, int kb) {
    Frag f;
    const __bf16* p = base + (size_t)row * ld + kb;
    f.h[0] = *reinterpret_cast<const bf16x8*>(p);
    f.h[1] = *reinterpret_cast<const bf16x8*>(p + 16);
    return f.v;
}

#define FLAG_ACC 1   // accumulate into existing C
#define FLAG_EPI 2   // bias + relu epilogue

// C[M,N] (+)= A[M,K](bf16) @ Bt[N,K](bf16)^T ; optional bias+relu; optional
// bf16 mirror of C. One wave = one 16x32 C tile. 8 waves per block.
__global__ __launch_bounds__(256)
void wmma_gemm_bf16(const __bf16* __restrict__ A, int lda,
                    const __bf16* __restrict__ Bt, int ldb,
                    float* __restrict__ C, __bf16* __restrict__ Cbf, int ldc,
                    const float* __restrict__ bias,
                    int M, int N, int K, int flags) {
    const int lane = threadIdx.x & 31;
    const int wave = threadIdx.x >> 5;
    const int tilesN = N >> 5;                  // 32-wide N tiles
    const int tilesM = (M + 15) >> 4;
    const int tile = blockIdx.x * 8 + wave;     // wave-uniform
    if (tile >= tilesM * tilesN) return;        // whole wave exits: EXEC stays full

    const int tm = tile / tilesN;
    const int tn = tile % tilesN;

    const int l15   = lane & 15;
    const int khalf = (lane >> 4) << 3;         // 0 or 8 (K sub-chunk select)

    // A operand row (clamped so OOB waves still load valid memory; stores guard)
    int rowA = tm * 16 + l15;
    if (rowA >= M) rowA = M - 1;

    // B operand rows = output columns (always in range: N multiple of 32)
    const int col0 = tn * 32 + l15;
    const int col1 = col0 + 16;

    // C layout: vgpr r -> m = mbase + r, n = ncol
    const int mbase = tm * 16 + ((lane >> 4) << 3);
    const int ncol0 = tn * 32 + l15;
    const int ncol1 = ncol0 + 16;

    f32x8 acc0 = {}, acc1 = {};
    if (flags & FLAG_ACC) {
#pragma unroll
        for (int r = 0; r < 8; ++r) {
            const int m = mbase + r;
            if (m < M) {
                acc0[r] = C[(size_t)m * ldc + ncol0];
                acc1[r] = C[(size_t)m * ldc + ncol1];
            }
        }
    }

    for (int k = 0; k < K; k += 32) {
        const bf16x16 a  = load_frag(A,  lda, rowA, k + khalf);
        const bf16x16 b0 = load_frag(Bt, ldb, col0, k + khalf);
        const bf16x16 b1 = load_frag(Bt, ldb, col1, k + khalf);
        acc0 = __builtin_amdgcn_wmma_f32_16x16x32_bf16(
            false, a, false, b0, (short)0, acc0, false, false);
        acc1 = __builtin_amdgcn_wmma_f32_16x16x32_bf16(
            false, a, false, b1, (short)0, acc1, false, false);
    }

    float bb0 = 0.f, bb1 = 0.f;
    if (flags & FLAG_EPI) { bb0 = bias[ncol0]; bb1 = bias[ncol1]; }

#pragma unroll
    for (int r = 0; r < 8; ++r) {
        const int m = mbase + r;
        if (m >= M) continue;
        float v0 = acc0[r], v1 = acc1[r];
        if (flags & FLAG_EPI) {
            v0 = fmaxf(v0 + bb0, 0.f);
            v1 = fmaxf(v1 + bb1, 0.f);
        }
        C[(size_t)m * ldc + ncol0] = v0;
        C[(size_t)m * ldc + ncol1] = v1;
        if (Cbf) {
            Cbf[(size_t)m * ldc + ncol0] = (__bf16)v0;
            Cbf[(size_t)m * ldc + ncol1] = (__bf16)v1;
        }
    }
}

// out[dst] += vals * h[src]  (COO scatter-add), 4 floats per thread.
__global__ void spmm_scatter(const float* __restrict__ vals,
                             const int* __restrict__ src,
                             const int* __restrict__ dst,
                             const float* __restrict__ h,
                             float* __restrict__ out,
                             int D, int nWork) {
    const int tid = blockIdx.x * blockDim.x + threadIdx.x;
    if (tid >= nWork) return;
    const int chunks = D >> 2;
    const int e = tid / chunks;
    const int c = (tid - e * chunks) << 2;
    const float w = vals[e];
    const float4 hv = *reinterpret_cast<const float4*>(h + (size_t)src[e] * D + c);
    float* o = out + (size_t)dst[e] * D + c;
    atomicAdd(o + 0, w * hv.x);
    atomicAdd(o + 1, w * hv.y);
    atomicAdd(o + 2, w * hv.z);
    atomicAdd(o + 3, w * hv.w);
}

__global__ void zero_f32(float* __restrict__ p, int n) {
    const int i = blockIdx.x * blockDim.x + threadIdx.x;
    if (i < n) p[i] = 0.f;
}

__global__ void cvt_f32_bf16(const float* __restrict__ in, __bf16* __restrict__ out, int n) {
    const int i = blockIdx.x * blockDim.x + threadIdx.x;
    if (i < n) out[i] = (__bf16)in[i];
}

// W[K,N] fp32 -> Wt[N,K] bf16 (weights; tiny)
__global__ void cvt_transpose_bf16(const float* __restrict__ in, __bf16* __restrict__ out,
                                   int K, int N) {
    const int i = blockIdx.x * blockDim.x + threadIdx.x;
    if (i >= K * N) return;
    const int n = i / K, k = i - n * K;
    out[(size_t)n * K + k] = (__bf16)in[(size_t)k * N + n];
}

// ---------------------------------------------------------------------------

extern "C" void kernel_launch(void* const* d_in, const int* in_sizes, int n_in,
                              void* d_out, int out_size, void* d_ws, size_t ws_size,
                              hipStream_t stream) {
    (void)in_sizes; (void)n_in; (void)out_size; (void)ws_size;

    const float* features = (const float*)d_in[0];
    const float* adj_vals = (const float*)d_in[1];
    const int*   adj_src  = (const int*)d_in[2];
    const int*   adj_dst  = (const int*)d_in[3];
    const float* W_gc1 = (const float*)d_in[4];   // 256x256
    const float* W_gc2 = (const float*)d_in[5];   // 256x128
    const float* W_gc3 = (const float*)d_in[6];   // 128x64
    const float* Wt1   = (const float*)d_in[7];   // 512x256
    const float* bt1   = (const float*)d_in[8];
    const float* Wt2   = (const float*)d_in[9];   // 384x128
    const float* bt2   = (const float*)d_in[10];
    const float* Wt3   = (const float*)d_in[11];  // 320x64
    const float* bt3   = (const float*)d_in[12];
    float* out = (float*)d_out;                   // [50000, 64]

    // workspace carve-up (256B aligned)
    char* ws = (char*)d_ws;
    size_t off = 0;
    auto alloc = [&](size_t bytes) -> void* {
        off = (off + 255) & ~(size_t)255;
        void* p = ws + off;
        off += bytes;
        return p;
    };
    __bf16* fbf  = (__bf16*)alloc((size_t)N_NODES * D_IN * 2);
    __bf16* wg1t = (__bf16*)alloc((size_t)H1 * D_IN * 2);         // [256][256]
    __bf16* wg2t = (__bf16*)alloc((size_t)H2 * H1 * 2);           // [128][256]
    __bf16* wg3t = (__bf16*)alloc((size_t)H3 * H2 * 2);           // [64][128]
    __bf16* wt1t = (__bf16*)alloc((size_t)H1 * (H1 + D_IN) * 2);  // [256][512]
    __bf16* wt2t = (__bf16*)alloc((size_t)H2 * (H2 + D_IN) * 2);  // [128][384]
    __bf16* wt3t = (__bf16*)alloc((size_t)H3 * (H3 + D_IN) * 2);  // [64][320]
    float*  Xf   = (float*)alloc((size_t)N_NODES * 256 * 4);      // x @ W_gc
    float*  Gf   = (float*)alloc((size_t)N_NODES * 256 * 4);      // spmm out
    __bf16* Gbf  = (__bf16*)alloc((size_t)N_NODES * 256 * 2);
    float*  Hf   = (float*)alloc((size_t)N_NODES * 256 * 4);      // hidden
    __bf16* Hbf  = (__bf16*)alloc((size_t)N_NODES * 256 * 2);

    const int M = N_NODES;
    auto blocks1d = [](int n) { return (n + 255) / 256; };
    auto gemm_blocks = [&](int N) { return (3125 * (N >> 5) + 7) / 8; };

    // --- precision conversion ------------------------------------------------
    cvt_f32_bf16<<<blocks1d(M * D_IN), 256, 0, stream>>>(features, fbf, M * D_IN);
    cvt_transpose_bf16<<<blocks1d(D_IN * H1), 256, 0, stream>>>(W_gc1, wg1t, D_IN, H1);
    cvt_transpose_bf16<<<blocks1d(H1 * H2), 256, 0, stream>>>(W_gc2, wg2t, H1, H2);
    cvt_transpose_bf16<<<blocks1d(H2 * H3), 256, 0, stream>>>(W_gc3, wg3t, H2, H3);
    cvt_transpose_bf16<<<blocks1d((H1 + D_IN) * H1), 256, 0, stream>>>(Wt1, wt1t, H1 + D_IN, H1);
    cvt_transpose_bf16<<<blocks1d((H2 + D_IN) * H2), 256, 0, stream>>>(Wt2, wt2t, H2 + D_IN, H2);
    cvt_transpose_bf16<<<blocks1d((H3 + D_IN) * H3), 256, 0, stream>>>(Wt3, wt3t, H3 + D_IN, H3);

    // --- layer 1 -------------------------------------------------------------
    // Xf = f @ W_gc1          [M,256]
    wmma_gemm_bf16<<<gemm_blocks(H1), 256, 0, stream>>>(
        fbf, D_IN, wg1t, D_IN, Xf, nullptr, H1, nullptr, M, H1, D_IN, 0);
    // Gf = spmm(Xf)
    zero_f32<<<blocks1d(M * H1), 256, 0, stream>>>(Gf, M * H1);
    spmm_scatter<<<blocks1d(N_EDGES * (H1 / 4)), 256, 0, stream>>>(
        adj_vals, adj_src, adj_dst, Xf, Gf, H1, N_EDGES * (H1 / 4));
    cvt_f32_bf16<<<blocks1d(M * H1), 256, 0, stream>>>(Gf, Gbf, M * H1);
    // Hf = relu(Gbf @ Wt1[0:256] + fbf @ Wt1[256:512] + bt1) ; Hbf mirror
    wmma_gemm_bf16<<<gemm_blocks(H1), 256, 0, stream>>>(
        Gbf, H1, wt1t, H1 + D_IN, Hf, nullptr, H1, nullptr, M, H1, H1, 0);
    wmma_gemm_bf16<<<gemm_blocks(H1), 256, 0, stream>>>(
        fbf, D_IN, wt1t + H1, H1 + D_IN, Hf, Hbf, H1, bt1, M, H1, D_IN,
        FLAG_ACC | FLAG_EPI);

    // --- layer 2 -------------------------------------------------------------
    wmma_gemm_bf16<<<gemm_blocks(H2), 256, 0, stream>>>(
        Hbf, H1, wg2t, H1, Xf, nullptr, H2, nullptr, M, H2, H1, 0);
    zero_f32<<<blocks1d(M * H2), 256, 0, stream>>>(Gf, M * H2);
    spmm_scatter<<<blocks1d(N_EDGES * (H2 / 4)), 256, 0, stream>>>(
        adj_vals, adj_src, adj_dst, Xf, Gf, H2, N_EDGES * (H2 / 4));
    cvt_f32_bf16<<<blocks1d(M * H2), 256, 0, stream>>>(Gf, Gbf, M * H2);
    wmma_gemm_bf16<<<gemm_blocks(H2), 256, 0, stream>>>(
        Gbf, H2, wt2t, H2 + D_IN, Hf, nullptr, H2, nullptr, M, H2, H2, 0);
    wmma_gemm_bf16<<<gemm_blocks(H2), 256, 0, stream>>>(
        fbf, D_IN, wt2t + H2, H2 + D_IN, Hf, Hbf, H2, bt2, M, H2, D_IN,
        FLAG_ACC | FLAG_EPI);

    // --- layer 3 -------------------------------------------------------------
    wmma_gemm_bf16<<<gemm_blocks(H3), 256, 0, stream>>>(
        Hbf, H2, wg3t, H2, Xf, nullptr, H3, nullptr, M, H3, H2, 0);
    zero_f32<<<blocks1d(M * H3), 256, 0, stream>>>(Gf, M * H3);
    spmm_scatter<<<blocks1d(N_EDGES * (H3 / 4)), 256, 0, stream>>>(
        adj_vals, adj_src, adj_dst, Xf, Gf, H3, N_EDGES * (H3 / 4));
    cvt_f32_bf16<<<blocks1d(M * H3), 256, 0, stream>>>(Gf, Gbf, M * H3);
    // emb = relu(Gbf @ Wt3[0:64] + fbf @ Wt3[64:320] + bt3)  -> d_out directly
    wmma_gemm_bf16<<<gemm_blocks(H3), 256, 0, stream>>>(
        Gbf, H3, wt3t, H3 + D_IN, out, nullptr, H3, nullptr, M, H3, H3, 0);
    wmma_gemm_bf16<<<gemm_blocks(H3), 256, 0, stream>>>(
        fbf, D_IN, wt3t + H3, H3 + D_IN, out, nullptr, H3, bt3, M, H3, D_IN,
        FLAG_ACC | FLAG_EPI);
}
